// TransformerBlock_43456479101581
// MI455X (gfx1250) — matrix operand order repaired and checked by reading, not compiled, not run on hardware
//
#include <hip/hip_runtime.h>
#include <cmath>

// ---------------------------------------------------------------------------
// Transformer block for MI455X (gfx1250, wave32, WMMA).
// All matmuls run through v_wmma_f32_16x16x32_bf16 (fp32 accumulate).
// ~137 GFLOP total vs <10us of HBM time at 23.3 TB/s => compute bound =>
// bf16 WMMA with software-pipelined (double-buffered) LDS staging.
// ---------------------------------------------------------------------------

#define D_MODEL 1024
#define N_HEADS 16
#define DK      64
#define D_FF    4096
#define SEQ     2048
#define BATCH   2
#define MROWS   (BATCH * SEQ)   // 4096 rows

typedef __bf16 bf16_t;
typedef __attribute__((ext_vector_type(16))) __bf16 v16bf;
typedef __attribute__((ext_vector_type(8)))  float  v8f;

union Frag { v16bf v; unsigned u[8]; };

__device__ __forceinline__ v8f wmma_bf16(const Frag& a, const Frag& b, v8f c) {
  // 8 args: (neg_a, A, neg_b, B, c_mod, C, reuse_a, reuse_b)
  return __builtin_amdgcn_wmma_f32_16x16x32_bf16(false, a.v, false, b.v,
                                                 (short)0, c, false, false);
}

// Row-wise reductions inside the two 16-lane halves (C-fragment N dimension).
__device__ __forceinline__ float rmax16(float v) {
  v = fmaxf(v, __shfl_xor(v, 1, 32));
  v = fmaxf(v, __shfl_xor(v, 2, 32));
  v = fmaxf(v, __shfl_xor(v, 4, 32));
  v = fmaxf(v, __shfl_xor(v, 8, 32));
  return v;
}
__device__ __forceinline__ float rsum16(float v) {
  v += __shfl_xor(v, 1, 32);
  v += __shfl_xor(v, 2, 32);
  v += __shfl_xor(v, 4, 32);
  v += __shfl_xor(v, 8, 32);
  return v;
}

// ---------------------------------------------------------------------------
// fp32 -> bf16 conversion (weights)
// ---------------------------------------------------------------------------
__global__ void cvt_f32_bf16(const float* __restrict__ in,
                             bf16_t* __restrict__ out, int n) {
  int i = blockIdx.x * 256 + threadIdx.x;
  if (i < n) out[i] = (bf16_t)in[i];
}

// ---------------------------------------------------------------------------
// LayerNorm over D_MODEL, fp32 in -> bf16 out. One block per row.
// ---------------------------------------------------------------------------
__global__ __launch_bounds__(256) void ln_bf16(const float* __restrict__ x,
                                               const float* __restrict__ g,
                                               const float* __restrict__ b,
                                               bf16_t* __restrict__ out) {
  __shared__ float red[256];
  const int row = blockIdx.x;
  const int t   = threadIdx.x;
  const float4 v = ((const float4*)(x + (size_t)row * D_MODEL))[t];

  red[t] = v.x + v.y + v.z + v.w;
  __syncthreads();
  for (int o = 128; o > 0; o >>= 1) { if (t < o) red[t] += red[t + o]; __syncthreads(); }
  const float mu = red[0] * (1.0f / D_MODEL);
  __syncthreads();

  const float dx = v.x - mu, dy = v.y - mu, dz = v.z - mu, dw = v.w - mu;
  red[t] = dx * dx + dy * dy + dz * dz + dw * dw;
  __syncthreads();
  for (int o = 128; o > 0; o >>= 1) { if (t < o) red[t] += red[t + o]; __syncthreads(); }
  const float rs = rsqrtf(red[0] * (1.0f / D_MODEL) + 1e-5f);

  const int c = t * 4;
  bf16_t* o4 = out + (size_t)row * D_MODEL + c;
  o4[0] = (bf16_t)(dx * rs * g[c + 0] + b[c + 0]);
  o4[1] = (bf16_t)(dy * rs * g[c + 1] + b[c + 1]);
  o4[2] = (bf16_t)(dz * rs * g[c + 2] + b[c + 2]);
  o4[3] = (bf16_t)(dw * rs * g[c + 3] + b[c + 3]);
}

// ---------------------------------------------------------------------------
// Tiled bf16 WMMA GEMM: C[M,N] = A[M,K] * B[K,N] + bias (+epilogue).
// Block tile 64x256, 8 waves of 32x64 (2x4 WMMA tiles), K chunk 32.
// Double-buffered LDS + register staging: global loads for chunk k+1 are in
// flight during the WMMAs of chunk k; one barrier per chunk.
// B staged transposed in LDS so fragment dwords are contiguous u32 reads.
// Rows padded to 34 elements (17 banks) => conflict-free fragment gathers.
// ---------------------------------------------------------------------------
enum { MODE_BF16 = 0, MODE_QSCALE = 1, MODE_GELU = 2, MODE_RESID = 3, MODE_FINAL = 4 };

template <int MODE>
__global__ __launch_bounds__(256) void gemm_bf16_k(
    const bf16_t* __restrict__ A, const bf16_t* __restrict__ Bw,
    const float* __restrict__ bias, const float* __restrict__ add,
    bf16_t* __restrict__ outH, float* __restrict__ outF, int K, int N) {
  __shared__ __align__(16) bf16_t As[2][64][34];
  __shared__ __align__(16) bf16_t Bs[2][256][34];

  const int tid  = threadIdx.x;
  const int m0   = blockIdx.y * 64;
  const int n0   = blockIdx.x * 256;
  const int w    = tid >> 5;
  const int lane = tid & 31;
  const int hlf  = lane >> 4;
  const int ln16 = lane & 15;
  const int wm   = w & 1;        // 2 waves along M
  const int wn   = w >> 1;       // 4 waves along N

  v8f acc[2][4] = {};
  const unsigned* Au = (const unsigned*)A;
  const unsigned* Bu = (const unsigned*)Bw;
  const int Ks2 = K >> 1, Ns2 = N >> 1;

  unsigned aReg[4], bReg[16];
  auto loadGlobal = [&](int kk) {
#pragma unroll
    for (int i = 0; i < 4; i++) {
      int idx = tid + i * 256;
      aReg[i] = Au[(size_t)(m0 + (idx >> 4)) * Ks2 + (kk >> 1) + (idx & 15)];
    }
#pragma unroll
    for (int i = 0; i < 16; i++) {
      int idx = tid + i * 256;
      bReg[i] = Bu[(size_t)(kk + (idx >> 7)) * Ns2 + (n0 >> 1) + (idx & 127)];
    }
  };
  auto storeLds = [&](int buf) {
#pragma unroll
    for (int i = 0; i < 4; i++) {
      int idx = tid + i * 256;
      *(unsigned*)&As[buf][idx >> 4][(idx & 15) * 2] = aReg[i];
    }
#pragma unroll
    for (int i = 0; i < 16; i++) {
      int idx = tid + i * 256;
      int r = idx >> 7, cp = idx & 127;
      Bs[buf][cp * 2 + 0][r] = ((const bf16_t*)&bReg[i])[0];
      Bs[buf][cp * 2 + 1][r] = ((const bf16_t*)&bReg[i])[1];
    }
  };

  const int nCh = K >> 5;
  loadGlobal(0);
  storeLds(0);
  __syncthreads();

  for (int c = 0; c < nCh; c++) {
    const int cur = c & 1, nxt = cur ^ 1;
    if (c + 1 < nCh) loadGlobal((c + 1) * 32);        // pipeline next chunk
    if (c + 2 < nCh) {                                // warm GL2 for chunk c+2
      __builtin_prefetch(&Au[(size_t)(m0 + (tid >> 4)) * Ks2 + ((c + 2) * 16) + (tid & 15)], 0, 3);
      __builtin_prefetch(&Bu[(size_t)((c + 2) * 32 + (tid >> 7)) * Ns2 + (n0 >> 1) + (tid & 127)], 0, 3);
    }

    // A fragments: row per lane, lanes>=16 take K=16..31.
    Frag af[2], bfr[4];
#pragma unroll
    for (int mt = 0; mt < 2; mt++) {
      int r  = wm * 32 + mt * 16 + ln16;
      int kb = hlf * 16;
#pragma unroll
      for (int j = 0; j < 8; j++) {
        int kp = kb + 2 * j;
        af[mt].u[j] = *(const unsigned*)&As[cur][r][kp];
      }
    }
    // B fragments: column per lane, lanes>=16 take K=16..31.
#pragma unroll
    for (int nt = 0; nt < 4; nt++) {
      int cc = wn * 64 + nt * 16 + ln16;
      int kb = hlf * 16;
#pragma unroll
      for (int j = 0; j < 8; j++)
        bfr[nt].u[j] = *(const unsigned*)&Bs[cur][cc][kb + 2 * j];
    }
#pragma unroll
    for (int mt = 0; mt < 2; mt++)
#pragma unroll
      for (int nt = 0; nt < 4; nt++)
        acc[mt][nt] = wmma_bf16(af[mt], bfr[nt], acc[mt][nt]);

    if (c + 1 < nCh) {
      storeLds(nxt);       // other buffer: safe while cur is still being read
      __syncthreads();
    }
  }

  // Epilogue. C layout: VGPR v -> row v + 8*(lane>=16), col = lane&15.
#pragma unroll
  for (int mt = 0; mt < 2; mt++) {
#pragma unroll
    for (int nt = 0; nt < 4; nt++) {
      int col = n0 + wn * 64 + nt * 16 + ln16;
      float bv = bias[col];
#pragma unroll
      for (int v = 0; v < 8; v++) {
        int row = m0 + wm * 32 + mt * 16 + v + 8 * hlf;
        size_t off = (size_t)row * N + col;
        float val = acc[mt][nt][v] + bv;
        if constexpr (MODE == MODE_BF16) {
          outH[off] = (bf16_t)val;
        } else if constexpr (MODE == MODE_QSCALE) {
          // Fold 1/sqrt(Dk)=0.125 into Q (exact power-of-two scale in bf16).
          outH[off] = (bf16_t)(val * 0.125f);
        } else if constexpr (MODE == MODE_GELU) {
          float gel = 0.5f * val * (1.0f + erff(val * 0.70710678118f));
          outH[off] = (bf16_t)gel;
        } else {  // MODE_RESID / MODE_FINAL: add residual, fp32 out
          outF[off] = val + add[off];
        }
      }
    }
  }
}

// ---------------------------------------------------------------------------
// Causal flash attention. Block = 4 waves x 16 query rows = 64 rows.
// Grid = B*H*(S/64). K/V chunks of 32 keys double-buffered in LDS
// (V transposed); global loads for chunk ch+1 overlap the WMMAs of chunk ch;
// one block barrier per chunk. Q arrives pre-scaled by 1/sqrt(Dk).
// All fragments are preloaded before their WMMAs so the scheduler can use
// partial dscnt waits. Causal masking runs only on the (wave-uniform)
// diagonal chunks; the steady-state path has no compares/selects.
// ---------------------------------------------------------------------------
__global__ __launch_bounds__(128) void attn_k(const bf16_t* __restrict__ Qg,
                                              const bf16_t* __restrict__ Kg,
                                              const bf16_t* __restrict__ Vg,
                                              bf16_t* __restrict__ Og) {
  __shared__ __align__(16) bf16_t Ks[2][32][66];     // [buf][key][d]
  __shared__ __align__(16) bf16_t Vt[2][64][34];     // [buf][d][key] (transposed)
  __shared__ __align__(16) bf16_t Pb[4][16][34];     // per-wave P tile

  const int tid  = threadIdx.x;
  const int w    = tid >> 5;
  const int lane = tid & 31;
  const int hlf  = lane >> 4;
  const int ln16 = lane & 15;

  const int idx = blockIdx.x;
  const int qb  = (idx & 31) * 64;          // S/64 = 32
  const int h   = (idx >> 5) & 15;
  const int b   = idx >> 9;
  const int sBase = b * SEQ;
  const int qRow0 = qb + w * 16;

  // Load Q A-fragments once (two K-chunks over d = 0..63).
  Frag qa[2];
  {
    const unsigned* Qu = (const unsigned*)Qg;
    size_t r = (size_t)(sBase + qRow0 + ln16) * (D_MODEL / 2);
    int kb = hlf * 16;
#pragma unroll
    for (int dc = 0; dc < 2; dc++)
#pragma unroll
      for (int j = 0; j < 8; j++) {
        int kp = kb + 2 * j;
        qa[dc].u[j] = Qu[r + (h * DK + dc * 32 + kp) / 2];
      }
  }

  v8f o[4] = {};
  float m[8], l[8];
#pragma unroll
  for (int v = 0; v < 8; v++) { m[v] = -INFINITY; l[v] = 0.0f; }

  const unsigned* Ku = (const unsigned*)Kg;
  const unsigned* Vu = (const unsigned*)Vg;
  const int nChunks = qb / 32 + 2;          // keys up to qb+63, uniform per block

  unsigned kReg[8], vReg[8];
  auto loadKV = [&](int t0) {
#pragma unroll
    for (int i = 0; i < 8; i++) {
      int id2 = tid + i * 128;
      size_t rowBase = (size_t)(sBase + t0 + (id2 >> 5)) * (D_MODEL / 2) + (h * DK) / 2;
      kReg[i] = Ku[rowBase + (id2 & 31)];
      vReg[i] = Vu[rowBase + (id2 & 31)];
    }
  };
  auto storeKV = [&](int buf) {
#pragma unroll
    for (int i = 0; i < 8; i++) {
      int id2 = tid + i * 128;
      int r = id2 >> 5, cp = id2 & 31;
      *(unsigned*)&Ks[buf][r][cp * 2] = kReg[i];
      Vt[buf][cp * 2 + 0][r] = ((const bf16_t*)&vReg[i])[0];
      Vt[buf][cp * 2 + 1][r] = ((const bf16_t*)&vReg[i])[1];
    }
  };

  loadKV(0);
  storeKV(0);
  __syncthreads();

  for (int ch = 0; ch < nChunks; ch++) {
    const int cur = ch & 1, nxt = cur ^ 1;
    const int t0 = ch * 32;
    if (ch + 1 < nChunks) loadKV((ch + 1) * 32);     // pipeline next chunk

    // Preload all four K B-fragments (2 key tiles x 2 d-chunks)...
    Frag kf[2][2];
#pragma unroll
    for (int kt = 0; kt < 2; kt++)
#pragma unroll
      for (int dc = 0; dc < 2; dc++) {
        int kb = hlf * 16;
#pragma unroll
        for (int j = 0; j < 8; j++)
          kf[kt][dc].u[j] =
              *(const unsigned*)&Ks[cur][kt * 16 + ln16][dc * 32 + kb + 2 * j];
      }
    // ...then issue the 4 WMMAs back-to-back (independent accumulators).
    v8f c0 = {}, c1 = {};
    c0 = wmma_bf16(qa[0], kf[0][0], c0);
    c1 = wmma_bf16(qa[0], kf[1][0], c1);
    c0 = wmma_bf16(qa[1], kf[0][1], c0);
    c1 = wmma_bf16(qa[1], kf[1][1], c1);

    // Online softmax update on C-fragment layout. t0/qRow0 are wave-uniform:
    // steady-state chunks are fully below the diagonal -> no masking code.
    float alpha[8];
    if (t0 + 31 <= qRow0) {                 // fast path: no causal mask
#pragma unroll
      for (int v = 0; v < 8; v++) {
        float s0 = c0[v], s1 = c1[v];
        float mt = rmax16(fmaxf(s0, s1));
        float mn = fmaxf(m[v], mt);
        alpha[v] = __expf(m[v] - mn);       // exp(-inf)=0 on first chunk
        float p0 = __expf(s0 - mn);
        float p1 = __expf(s1 - mn);
        l[v] = l[v] * alpha[v] + rsum16(p0 + p1);
        m[v] = mn;
        Pb[w][v + 8 * hlf][ln16]      = (bf16_t)p0;
        Pb[w][v + 8 * hlf][16 + ln16] = (bf16_t)p1;
      }
    } else {                                // diagonal chunks: mask to -inf
#pragma unroll
      for (int v = 0; v < 8; v++) {
        int row  = qRow0 + v + 8 * hlf;
        int col0 = t0 + ln16, col1 = t0 + 16 + ln16;
        float s0 = c0[v]; if (col0 > row) s0 = -INFINITY;
        float s1 = c1[v]; if (col1 > row) s1 = -INFINITY;
        float mt = rmax16(fmaxf(s0, s1));
        float mn = fmaxf(m[v], mt);
        alpha[v] = (mn == -INFINITY) ? 1.0f : __expf(m[v] - mn);
        float p0 = (s0 == -INFINITY) ? 0.0f : __expf(s0 - mn);
        float p1 = (s1 == -INFINITY) ? 0.0f : __expf(s1 - mn);
        l[v] = l[v] * alpha[v] + rsum16(p0 + p1);
        m[v] = mn;
        Pb[w][v + 8 * hlf][ln16]      = (bf16_t)p0;
        Pb[w][v + 8 * hlf][16 + ln16] = (bf16_t)p1;
      }
    }
#pragma unroll
    for (int dt = 0; dt < 4; dt++)
#pragma unroll
      for (int v = 0; v < 8; v++) o[dt][v] *= alpha[v];

    // Wave-private LDS round-trip: LDS is in-order per wave; just stop the
    // compiler from reordering the fragment loads above the stores.
    __builtin_amdgcn_wave_barrier();

    // Preload P A-fragment and all four V B-fragments, then 4 WMMAs.
    Frag pa, vf[4];
    {
      int kb = hlf * 16;
#pragma unroll
      for (int j = 0; j < 8; j++) {
        int kp = kb + 2 * j;
        pa.u[j] = *(const unsigned*)&Pb[w][ln16][kp];
      }
    }
#pragma unroll
    for (int dt = 0; dt < 4; dt++) {
      int kb = hlf * 16;
#pragma unroll
      for (int j = 0; j < 8; j++)
        vf[dt].u[j] = *(const unsigned*)&Vt[cur][dt * 16 + ln16][kb + 2 * j];
    }
#pragma unroll
    for (int dt = 0; dt < 4; dt++) o[dt] = wmma_bf16(pa, vf[dt], o[dt]);

    if (ch + 1 < nChunks) {
      storeKV(nxt);        // other buffer: safe while cur is still being read
      __syncthreads();
    }
  }

  // Normalize and write merged-head attn_out (bf16) for the Wo GEMM.
#pragma unroll
  for (int v = 0; v < 8; v++) {
    float invl = 1.0f / l[v];
    int s = qRow0 + v + 8 * hlf;
    size_t base = (size_t)(sBase + s) * D_MODEL + h * DK;
#pragma unroll
    for (int dt = 0; dt < 4; dt++)
      Og[base + dt * 16 + ln16] = (bf16_t)(o[dt][v] * invl);
  }
}

// ---------------------------------------------------------------------------
// Host-side orchestration (graph-capture safe; everything on `stream`).
// ---------------------------------------------------------------------------
extern "C" void kernel_launch(void* const* d_in, const int* in_sizes, int n_in,
                              void* d_out, int out_size, void* d_ws, size_t ws_size,
                              hipStream_t stream) {
  const float* x    = (const float*)d_in[0];
  // d_in[1] = mask: causal, recomputed analytically in-kernel.
  const float* ln1g = (const float*)d_in[2];
  const float* ln1b = (const float*)d_in[3];
  const float* ln2g = (const float*)d_in[4];
  const float* ln2b = (const float*)d_in[5];
  const float* Wq = (const float*)d_in[6];   const float* bq = (const float*)d_in[7];
  const float* Wk = (const float*)d_in[8];   const float* bk = (const float*)d_in[9];
  const float* Wv = (const float*)d_in[10];  const float* bv = (const float*)d_in[11];
  const float* Wo = (const float*)d_in[12];  const float* bo = (const float*)d_in[13];
  const float* W1 = (const float*)d_in[14];  const float* b1 = (const float*)d_in[15];
  const float* W2 = (const float*)d_in[16];  const float* b2 = (const float*)d_in[17];

  char* ws = (char*)d_ws;
  size_t off = 0;
  auto alloc = [&](size_t bytes) -> void* {
    void* p = ws + off;
    off += (bytes + 255) & ~(size_t)255;
    return p;
  };
  bf16_t* wqh   = (bf16_t*)alloc((size_t)D_MODEL * D_MODEL * 2);
  bf16_t* wkh   = (bf16_t*)alloc((size_t)D_MODEL * D_MODEL * 2);
  bf16_t* wvh   = (bf16_t*)alloc((size_t)D_MODEL * D_MODEL * 2);
  bf16_t* woh   = (bf16_t*)alloc((size_t)D_MODEL * D_MODEL * 2);
  bf16_t* w1h   = (bf16_t*)alloc((size_t)D_MODEL * D_FF * 2);
  bf16_t* w2h   = (bf16_t*)alloc((size_t)D_FF * D_MODEL * 2);
  bf16_t* n1h   = (bf16_t*)alloc((size_t)MROWS * D_MODEL * 2);
  bf16_t* Qh    = (bf16_t*)alloc((size_t)MROWS * D_MODEL * 2);
  bf16_t* Kh    = (bf16_t*)alloc((size_t)MROWS * D_MODEL * 2);
  bf16_t* Vh    = (bf16_t*)alloc((size_t)MROWS * D_MODEL * 2);
  bf16_t* atth  = (bf16_t*)alloc((size_t)MROWS * D_MODEL * 2);
  float*  res   = (float*) alloc((size_t)MROWS * D_MODEL * 4);
  bf16_t* n2h   = (bf16_t*)alloc((size_t)MROWS * D_MODEL * 2);
  bf16_t* ffh   = (bf16_t*)alloc((size_t)MROWS * D_FF * 2);
  // total ~120 MB of d_ws

  const int nW = D_MODEL * D_MODEL, nF = D_MODEL * D_FF;
  cvt_f32_bf16<<<(nW + 255) / 256, 256, 0, stream>>>(Wq, wqh, nW);
  cvt_f32_bf16<<<(nW + 255) / 256, 256, 0, stream>>>(Wk, wkh, nW);
  cvt_f32_bf16<<<(nW + 255) / 256, 256, 0, stream>>>(Wv, wvh, nW);
  cvt_f32_bf16<<<(nW + 255) / 256, 256, 0, stream>>>(Wo, woh, nW);
  cvt_f32_bf16<<<(nF + 255) / 256, 256, 0, stream>>>(W1, w1h, nF);
  cvt_f32_bf16<<<(nF + 255) / 256, 256, 0, stream>>>(W2, w2h, nF);

  ln_bf16<<<MROWS, 256, 0, stream>>>(x, ln1g, ln1b, n1h);

  dim3 gD(D_MODEL / 256, MROWS / 64);   // (4, 64)
  gemm_bf16_k<MODE_QSCALE><<<gD, 256, 0, stream>>>(n1h, wqh, bq, nullptr, Qh, nullptr, D_MODEL, D_MODEL);
  gemm_bf16_k<MODE_BF16><<<gD, 256, 0, stream>>>(n1h, wkh, bk, nullptr, Kh, nullptr, D_MODEL, D_MODEL);
  gemm_bf16_k<MODE_BF16><<<gD, 256, 0, stream>>>(n1h, wvh, bv, nullptr, Vh, nullptr, D_MODEL, D_MODEL);

  attn_k<<<BATCH * N_HEADS * (SEQ / 64), 128, 0, stream>>>(Qh, Kh, Vh, atth);

  gemm_bf16_k<MODE_RESID><<<gD, 256, 0, stream>>>(atth, woh, bo, x, nullptr, res, D_MODEL, D_MODEL);

  ln_bf16<<<MROWS, 256, 0, stream>>>(res, ln2g, ln2b, n2h);

  dim3 gF(D_FF / 256, MROWS / 64);      // (16, 64)
  gemm_bf16_k<MODE_GELU><<<gF, 256, 0, stream>>>(n2h, w1h, b1, nullptr, ffh, nullptr, D_MODEL, D_FF);
  gemm_bf16_k<MODE_FINAL><<<gD, 256, 0, stream>>>(ffh, w2h, b2, res, nullptr, (float*)d_out, D_FF, D_MODEL);
}
